// Lstm_22668837388923
// MI455X (gfx1250) — matrix operand order repaired and checked by reading, not compiled, not run on hardware
//
#include <hip/hip_runtime.h>

// ---------------------------------------------------------------------------
// Persistent-cluster LSTM for MI455X (gfx1250, wave32, WMMA).
//
//   grid = 64 workgroups = 4 "clusters" x 16 WGs (cluster barrier syncs the
//   16 WGs each timestep; batch rows are independent across clusters).
//   cluster c  : batch rows [16c, 16c+16)          (M = 16 = WMMA tile)
//   WG w in cl : hidden cols [32w, 32w+32)  -> 128 gate columns (4 gates x 32)
//
//   Per step, per WG:  fused GEMM  [xe_t | h_{t-1}] (16x768) @ Wc^T (768x128)
//   with Wc = [W_ih | W_hh] slice resident in LDS as bf16 for all 512 steps.
//   8 waves, one 16x16 tile each, 24 x v_wmma_f32_16x16x32_bf16 per step.
//   h panel staged with global_load_async_to_lds_b128 (ASYNCcnt).
// ---------------------------------------------------------------------------

typedef __attribute__((ext_vector_type(16))) __bf16 v16bf;
typedef __attribute__((ext_vector_type(8)))  __bf16 v8bf;
typedef __attribute__((ext_vector_type(8)))  float  v8f;

#define B_    64
#define S_    512
#define D_    256
#define H_    512
#define KTOT  768          // D_ + H_ (fused K)
#define ROWS  16           // batch rows per cluster
#define NCOL  128          // gate columns per WG (4 gates x 32)
#define HC    32           // hidden columns per WG

// Raw-HW sigmoid / tanh: v_exp_f32 is base-2, v_rcp_f32 without div fixup.
__device__ __forceinline__ float fast_sigmoid(float x) {
  return __builtin_amdgcn_rcpf(1.0f + __builtin_amdgcn_exp2f(-1.4426950408889634f * x));
}
__device__ __forceinline__ float fast_tanh(float x) {
  return 2.0f * __builtin_amdgcn_rcpf(1.0f + __builtin_amdgcn_exp2f(-2.8853900817779268f * x)) - 1.0f;
}

__global__ void __launch_bounds__(256)
lstm_init_h(const float* __restrict__ h0, __bf16* __restrict__ hbuf) {
  int i = blockIdx.x * 256 + threadIdx.x;
  if (i < B_ * H_) hbuf[i] = (__bf16)h0[i];
}

__global__ void __launch_bounds__(256)
lstm_persistent(const int*   __restrict__ x,      // (B,S)
                const float* __restrict__ emb,    // (V,D)
                const float* __restrict__ W_ih,   // (4H,D)
                const float* __restrict__ W_hh,   // (4H,H)
                const float* __restrict__ b_ih,   // (4H)
                const float* __restrict__ b_hh,   // (4H)
                const float* __restrict__ c0,     // (B,H)
                __bf16*      __restrict__ hbuf,   // (2,B,H) bf16 double buffer
                float*       __restrict__ out)    // (S,B,H) f32
{
  __shared__ __bf16 lds_w[NCOL * KTOT];   // 192 KB  weight slice (resident)
  __shared__ __bf16 lds_a[ROWS * KTOT];   //  24 KB  A panel [xe_t | h_{t-1}]
  __shared__ float  lds_g[4 * ROWS * HC]; //   8 KB  gate tiles (i,f,g,o)
  __shared__ float  lds_c[ROWS * HC];     //   2 KB  cell state (resident)
  __shared__ float  lds_bias[NCOL];       //  .5 KB  b_ih + b_hh

  const int tid = threadIdx.x;
  const int wg  = blockIdx.x;     // 0..63
  const int cl  = wg >> 4;        // cluster (batch chunk)
  const int wic = wg & 15;        // WG-in-cluster (hidden chunk)
  const int m0  = cl * ROWS;      // first batch row
  const int hc0 = wic * HC;       // first hidden column

  // ---- one-time: stage weight slice + bias + cell state into LDS ----------
  {
    const int n    = tid >> 1;              // 0..127 (gate column within WG)
    const int half = tid & 1;               // K half: [0,384) or [384,768)
    const int grow = (n >> 5) * H_ + hc0 + (n & 31);   // row in W_ih / W_hh
    const float* wih = W_ih + (size_t)grow * D_;
    const float* whh = W_hh + (size_t)grow * H_;
    const int kbeg = half * (KTOT / 2);
    for (int k = kbeg; k < kbeg + KTOT / 2; k += 4) {
      float4 v = (k < D_) ? *(const float4*)(wih + k)
                          : *(const float4*)(whh + (k - D_));
      __bf16* dst = &lds_w[n * KTOT + k];
      dst[0] = (__bf16)v.x; dst[1] = (__bf16)v.y;
      dst[2] = (__bf16)v.z; dst[3] = (__bf16)v.w;
    }
    if (tid < NCOL) {
      int gr = (tid >> 5) * H_ + hc0 + (tid & 31);
      lds_bias[tid] = b_ih[gr] + b_hh[gr];
    }
  }
  for (int e = tid; e < ROWS * HC; e += 256) {
    int m = e >> 5, j = e & 31;
    lds_c[e] = c0[(size_t)(m0 + m) * H_ + hc0 + j];
  }
  __syncthreads();

  // ---- wave / lane geometry for WMMA operand builds -----------------------
  const int wave = tid >> 5;            // 0..7 : 16-col N tile per wave
  const int lane = tid & 31;
  const int ln   = lane & 15;           // N (B/C/D) or M (A) index
  const int hi   = lane >> 4;           // lane-half selects K sub-block
  const int n0   = wave * 16;           // first gate column of this wave
  const int gate = wave >> 1;           // 0:i 1:f 2:g 3:o
  const int csub = (wave & 1) * 16;     // column offset within gate's 32

  for (int t = 0; t < S_; ++t) {
    const __bf16* hread  = hbuf + (size_t)( t      & 1) * B_ * H_;
    __bf16*       hwrite = hbuf + (size_t)((t + 1) & 1) * B_ * H_;

    // ---- stage A panel ----------------------------------------------------
    {
      const int mb  = tid >> 4;   // batch row 0..15
      const int seg = tid & 15;   // 16-wide chunk

      // h_{t-1}: raw bf16, CDNA5 async copy global -> LDS (no VGPR bounce)
      {
        const __bf16* src = hread + (size_t)(m0 + mb) * H_ + seg * 32;
        unsigned lds_off  = (unsigned)(uintptr_t)&lds_a[mb * KTOT + D_ + seg * 32];
        unsigned long long ga = (unsigned long long)(uintptr_t)src;
        #pragma unroll
        for (int q = 0; q < 4; ++q) {
          asm volatile("global_load_async_to_lds_b128 %0, %1, off"
                       :: "v"(lds_off + q * 16), "v"(ga + q * 16) : "memory");
        }
      }

      // xe_t: gather embedding row, convert f32 -> bf16 in VGPRs
      const int idx = x[(m0 + mb) * S_ + t];
      const float* er = emb + (size_t)idx * D_ + seg * 16;
      __bf16* dst = &lds_a[mb * KTOT + seg * 16];
      #pragma unroll
      for (int q = 0; q < 4; ++q) {
        float4 v = *(const float4*)(er + q * 4);
        dst[q * 4 + 0] = (__bf16)v.x; dst[q * 4 + 1] = (__bf16)v.y;
        dst[q * 4 + 2] = (__bf16)v.z; dst[q * 4 + 3] = (__bf16)v.w;
      }
      if (t + 1 < S_) {   // warm L2 for next step's gather (global_prefetch_b8)
        int nidx = x[(m0 + mb) * S_ + t + 1];
        __builtin_prefetch(emb + (size_t)nidx * D_ + seg * 16, 0, 1);
      }
    }
    asm volatile("s_wait_asynccnt 0" ::: "memory");  // async h-panel landed
    __syncthreads();

    // ---- GEMM: 16x16 tile per wave, K = 768, bf16 WMMA, f32 accum ---------
    v8f acc;
    {
      float bb = lds_bias[n0 + ln];            // bias depends on N only
      #pragma unroll
      for (int r = 0; r < 8; ++r) acc[r] = bb;
    }
    const __bf16* arow = &lds_a[ln * KTOT];          // A: row M = ln
    const __bf16* brow = &lds_w[(n0 + ln) * KTOT];   // B: col N = ln ([N][K] row-major)
    #pragma unroll 4
    for (int k0 = 0; k0 < KTOT; k0 += 32) {
      // A 16x32 bf16: lane half hi -> K = hi*8 + [0,8) and 16 + hi*8 + [0,8)
      v8bf a_lo = *(const v8bf*)(arow + k0 + hi * 8);
      v8bf a_hi = *(const v8bf*)(arow + k0 + 16 + hi * 8);
      // B 32x16 bf16: lane half hi -> K = hi*8 + [0,8) and 16 + hi*8 + [0,8)
      v8bf b_lo = *(const v8bf*)(brow + k0 + hi * 8);
      v8bf b_hi = *(const v8bf*)(brow + k0 + 16 + hi * 8);
      v16bf av, bv;
      #pragma unroll
      for (int q = 0; q < 8; ++q) {
        av[q] = a_lo[q]; av[8 + q] = a_hi[q];
        bv[q] = b_lo[q]; bv[8 + q] = b_hi[q];
      }
      acc = __builtin_amdgcn_wmma_f32_16x16x32_bf16(
                false, av, false, bv, (short)0, acc, false, false);
    }

    // ---- scatter gate tile to LDS (C layout: m = r + 8*hi, n = ln) --------
    #pragma unroll
    for (int r = 0; r < 8; ++r) {
      int m = r + hi * 8;
      lds_g[(gate * ROWS + m) * HC + csub + ln] = acc[r];
    }
    __syncthreads();

    // ---- pointwise LSTM cell (f32, raw HW transcendentals), emit h --------
    #pragma unroll
    for (int u = 0; u < 2; ++u) {
      int e = tid + u * 256;                 // ROWS*HC = 512 elements
      int m = e >> 5, j = e & 31;
      float gi = lds_g[(0 * ROWS + m) * HC + j];
      float gf = lds_g[(1 * ROWS + m) * HC + j];
      float gg = lds_g[(2 * ROWS + m) * HC + j];
      float go = lds_g[(3 * ROWS + m) * HC + j];
      float c  = fast_sigmoid(gf) * lds_c[e] + fast_sigmoid(gi) * fast_tanh(gg);
      float h  = fast_sigmoid(go) * fast_tanh(c);
      lds_c[e] = c;
      out[(size_t)t * B_ * H_ + (size_t)(m0 + m) * H_ + hc0 + j] = h;  // hs(S,B,H)
      hwrite[(size_t)(m0 + m) * H_ + hc0 + j] = (__bf16)h;
    }
    // Stores land in L2 (device-visible) once STOREcnt drains — cheap fence.
    asm volatile("s_wait_storecnt 0" ::: "memory");
    __syncthreads();
    __builtin_amdgcn_s_cluster_barrier();   // s_barrier_signal/wait -3
  }
}

extern "C" void kernel_launch(void* const* d_in, const int* in_sizes, int n_in,
                              void* d_out, int out_size, void* d_ws, size_t ws_size,
                              hipStream_t stream) {
  (void)in_sizes; (void)n_in; (void)out_size; (void)ws_size;
  const int*   x    = (const int*)  d_in[0];
  const float* emb  = (const float*)d_in[1];
  const float* W_ih = (const float*)d_in[2];
  const float* W_hh = (const float*)d_in[3];
  const float* b_ih = (const float*)d_in[4];
  const float* b_hh = (const float*)d_in[5];
  const float* h0   = (const float*)d_in[6];
  const float* c0   = (const float*)d_in[7];
  float*  out  = (float*)d_out;
  __bf16* hbuf = (__bf16*)d_ws;   // 2 * 64 * 512 bf16 = 128 KB double buffer

  lstm_init_h<<<(B_ * H_ + 255) / 256, 256, 0, stream>>>(h0, hbuf);
  lstm_persistent<<<64, 256, 0, stream>>>(x, emb, W_ih, W_hh, b_ih, b_hh, c0,
                                          hbuf, out);
}
